// WCE_J_SIMPL_58291296141874
// MI455X (gfx1250) — matrix operand order repaired and checked by reading, not compiled, not run on hardware
//
#include <hip/hip_runtime.h>
#include <hip/hip_fp16.h>

typedef __attribute__((ext_vector_type(16))) _Float16 v16h;
typedef __attribute__((ext_vector_type(8)))  _Float16 v8h;
typedef __attribute__((ext_vector_type(8)))  float    v8f;

#define NB      8
#define NCLS    21
#define HWSZ    (512 * 512)              // 2^18
#define NPIX    (NB * HWSZ)              // 2,097,152
#define GRID_X  1024
#define BLOCK_X 256
#define ITERS   (NPIX / (GRID_X * BLOCK_X))   // == 8 exactly

static_assert(NPIX % (GRID_X * BLOCK_X) == 0, "exact division required");

// ---------------------------------------------------------------------------
// Kernel 0: zero the workspace accumulators (harness poisons d_ws with 0xAA).
// ws layout (floats): [0..1023] S (32x32), [1024..1055] counts, [1056] logpy.
// ---------------------------------------------------------------------------
__global__ void mcc_zero_ws(float* __restrict__ ws) {
  const int i = blockIdx.x * blockDim.x + threadIdx.x;
  if (i < 1057) ws[i] = 0.0f;
}

// ---------------------------------------------------------------------------
// Kernel 1: streaming pass. Per pixel: target-argmax, softmax, logpy.
// Per wave: accumulate S += P(21x32,f16) * onehot(32x21,f16) via
// v_wmma_f32_16x16x32_f16 (2x2 tiles of 16x16, padded to 32x32).
// ---------------------------------------------------------------------------
__global__ __launch_bounds__(BLOCK_X) void mcc_main_kernel(
    const float* __restrict__ inp, const float* __restrict__ tgt,
    float* __restrict__ gS, float* __restrict__ gCnt, float* __restrict__ gLog)
{
  __shared__ __align__(16) _Float16      Pbuf[8][32 * 32];   // per-wave P staging (class-major)
  __shared__ __align__(16) unsigned char Lbuf[8][32];        // per-wave labels
  __shared__ float sS[32 * 32];                              // block-level S partial
  __shared__ float sCnt[32];
  __shared__ float sLog;

  const int tid  = threadIdx.x;
  const int lane = tid & 31;
  const int wav  = tid >> 5;
  const int hsel = lane >> 4;        // lane half-select (wave32 fragment layouts)
  const int r0   = lane & 15;

  for (int i = tid; i < 32 * 32; i += BLOCK_X) sS[i] = 0.0f;
  if (tid < 32) sCnt[tid] = 0.0f;
  if (tid == 0) sLog = 0.0f;
  // zero-pad classes 21..31 in this wave's staging tile (written only here)
  for (int r = NCLS; r < 32; ++r) Pbuf[wav][r * 32 + lane] = (_Float16)0.0f;

  v8f c00 = {}, c01 = {}, c10 = {}, c11 = {};
  float logAcc = 0.0f;

  const int gid = blockIdx.x * BLOCK_X + tid;
  for (int it = 0; it < ITERS; ++it) {
    const int pix  = gid + it * (GRID_X * BLOCK_X);
    const int n    = pix >> 18;
    const int hw   = pix & (HWSZ - 1);
    const int base = ((n * NCLS) << 18) + hw;

    // ---- argmax over target channels (first max wins, like jnp.argmax) ----
    float tmax = __builtin_nontemporal_load(&tgt[base]);
    int label = 0;
#pragma unroll
    for (int c = 1; c < NCLS; ++c) {
      const float tv = __builtin_nontemporal_load(&tgt[base + (c << 18)]);
      if (tv > tmax) { tmax = tv; label = c; }
    }

    // ---- softmax / log-softmax over input channels ----
    float x[NCLS];
#pragma unroll
    for (int c = 0; c < NCLS; ++c)
      x[c] = __builtin_nontemporal_load(&inp[base + (c << 18)]);
    float m = x[0];
#pragma unroll
    for (int c = 1; c < NCLS; ++c) m = fmaxf(m, x[c]);
    float xl = x[0];
#pragma unroll
    for (int c = 1; c < NCLS; ++c) if (c == label) xl = x[c];
    float s = 0.0f;
#pragma unroll
    for (int c = 0; c < NCLS; ++c) { x[c] = __expf(x[c] - m); s += x[c]; }
    const float inv = 1.0f / s;
    logAcc += (xl - m) - __logf(s);      // logp at the label

    // ---- stage p (f16) and labels for this wave's 32-pixel batch ----
#pragma unroll
    for (int c = 0; c < NCLS; ++c)
      Pbuf[wav][c * 32 + lane] = (_Float16)(x[c] * inv);
    Lbuf[wav][lane] = (unsigned char)label;
    atomicAdd(&sCnt[label], 1.0f);       // exact integer-valued counts

    __syncthreads();

    // ---- build A fragments (16-bit A 16x32 layout) ----
    const _Float16* Pw = Pbuf[wav];
    const int ka = hsel * 8;             // lanes 0-15: K 0-7/16-23; lanes 16-31: K 8-15/24-31
    v16h a0, a1;
    {
      v8h lo = *(const v8h*)(Pw + r0 * 32 + ka);
      v8h hi = *(const v8h*)(Pw + r0 * 32 + 16 + ka);
#pragma unroll
      for (int t = 0; t < 8; ++t) { a0[t] = lo[t]; a0[8 + t] = hi[t]; }
      lo = *(const v8h*)(Pw + (16 + r0) * 32 + ka);
      hi = *(const v8h*)(Pw + (16 + r0) * 32 + 16 + ka);
#pragma unroll
      for (int t = 0; t < 8; ++t) { a1[t] = lo[t]; a1[8 + t] = hi[t]; }
    }

    // ---- build one-hot B fragments (32x16: lanes 0-15 K 0-7/16-23, lanes 16-31 K 8-15/24-31) ----
    const unsigned int* lw = (const unsigned int*)(&Lbuf[wav][hsel * 8]);
    const unsigned int w0 = lw[0], w1 = lw[1], w2 = lw[4], w3 = lw[5];
    int labk[16];
#pragma unroll
    for (int t = 0; t < 4; ++t) {
      labk[t]      = (int)((w0 >> (8 * t)) & 0xFF);
      labk[4 + t]  = (int)((w1 >> (8 * t)) & 0xFF);
      labk[8 + t]  = (int)((w2 >> (8 * t)) & 0xFF);
      labk[12 + t] = (int)((w3 >> (8 * t)) & 0xFF);
    }
    v16h b0, b1;
#pragma unroll
    for (int t = 0; t < 16; ++t) {
      b0[t] = (labk[t] == r0)      ? (_Float16)1.0f : (_Float16)0.0f;
      b1[t] = (labk[t] == 16 + r0) ? (_Float16)1.0f : (_Float16)0.0f;
    }

    // ---- S_tile += A x B (the 'nihw,njhw->ij' einsum), f32 accumulate ----
    c00 = __builtin_amdgcn_wmma_f32_16x16x32_f16(false, a0, false, b0, (short)0, c00, false, false);
    c01 = __builtin_amdgcn_wmma_f32_16x16x32_f16(false, a0, false, b1, (short)0, c01, false, false);
    c10 = __builtin_amdgcn_wmma_f32_16x16x32_f16(false, a1, false, b0, (short)0, c10, false, false);
    c11 = __builtin_amdgcn_wmma_f32_16x16x32_f16(false, a1, false, b1, (short)0, c11, false, false);

    __syncthreads();   // WAR: next iteration re-stages Pbuf/Lbuf
  }

  // ---- reduce wave accumulators into block partial (C/D layout: VGPR r -> M=r / r+8) ----
#pragma unroll
  for (int r = 0; r < 8; ++r) {
    const int rr = r + (hsel ? 8 : 0);
    atomicAdd(&sS[rr * 32 + r0],             c00[r]);
    atomicAdd(&sS[rr * 32 + 16 + r0],        c01[r]);
    atomicAdd(&sS[(16 + rr) * 32 + r0],      c10[r]);
    atomicAdd(&sS[(16 + rr) * 32 + 16 + r0], c11[r]);
  }
  atomicAdd(&sLog, logAcc);
  __syncthreads();

  // ---- one flush of block partials to global accumulators ----
  for (int i = tid; i < 32 * 32; i += BLOCK_X) {
    const int ii = i >> 5, jj = i & 31;
    if (ii < NCLS && jj < NCLS) atomicAdd(&gS[i], sS[i]);
  }
  if (tid < NCLS) atomicAdd(&gCnt[tid], sCnt[tid]);
  if (tid == 0) atomicAdd(gLog, sLog);
}

// ---------------------------------------------------------------------------
// Kernel 2: 21x21 epilogue -> scalar loss.
// ---------------------------------------------------------------------------
__global__ void mcc_finalize(const float* __restrict__ gS,
                             const float* __restrict__ gCnt,
                             const float* __restrict__ gLog,
                             float* __restrict__ out) {
  __shared__ float acc;
  if (threadIdx.x == 0) acc = 0.0f;
  __syncthreads();
  float part = 0.0f;
  for (int p = threadIdx.x; p < NCLS * NCLS; p += blockDim.x) {
    const int i = p / NCLS, j = p % NCLS;
    const float ci = gCnt[i], cj = gCnt[j];
    if (i != j && ci > 0.0f && cj > 0.0f) {
      const float mcc = gS[i * 32 + i] / ci - gS[i * 32 + j] / cj;
      part += -0.5f * logf((0.5f * mcc + 0.5f) + 1e-8f);   // POWER=1, lamb=0.5
    }
  }
  atomicAdd(&acc, part);
  __syncthreads();
  if (threadIdx.x == 0)
    out[0] = acc - gLog[0] * (1.0f / (float)NPIX);          // lossd + (-logpy.mean())
}

// ---------------------------------------------------------------------------
extern "C" void kernel_launch(void* const* d_in, const int* in_sizes, int n_in,
                              void* d_out, int out_size, void* d_ws, size_t ws_size,
                              hipStream_t stream) {
  (void)in_sizes; (void)n_in; (void)out_size; (void)ws_size;
  const float* inp = (const float*)d_in[0];
  const float* tgt = (const float*)d_in[1];
  float* ws   = (float*)d_ws;
  float* gS   = ws;            // 32*32
  float* gCnt = ws + 1024;     // 32
  float* gLog = ws + 1056;     // 1
  float* out  = (float*)d_out;

  mcc_zero_ws<<<5, 256, 0, stream>>>(ws);
  mcc_main_kernel<<<GRID_X, BLOCK_X, 0, stream>>>(inp, tgt, gS, gCnt, gLog);
  mcc_finalize<<<1, 256, 0, stream>>>(gS, gCnt, gLog, out);
}
